// Swin_transformer_22952305229914
// MI455X (gfx1250) — matrix operand order repaired and checked by reading, not compiled, not run on hardware
//
#include <hip/hip_runtime.h>
#include <cmath>

#define B_     64
#define HS_    56
#define WS_    56
#define C_     128
#define WIN_   7
#define SHIFT_ 3
#define NH_    4
#define DH_    32
#define N_     49
#define NWH_   8
#define NW_    64
#define HID_   512
#define NTOK   (B_*HS_*WS_)     /* 200704 */
#define NWINT  (B_*NW_)         /* 4096   */
#define OH_    28
#define MTOK   (B_*OH_*OH_)     /* 50176  */

typedef __bf16 bf16;
typedef __attribute__((ext_vector_type(16))) __bf16 v16bf;
typedef __attribute__((ext_vector_type(8)))  __bf16 v8bf;
typedef __attribute__((ext_vector_type(8)))  float  v8f;
typedef __attribute__((ext_vector_type(4)))  unsigned int v4u;
typedef __attribute__((ext_vector_type(8)))  int v8i;
typedef __attribute__((ext_vector_type(4)))  int v4i;

union V16U { v16bf v; v8bf h[2]; };

__device__ __forceinline__ bf16 tobf(float x) { return (bf16)x; }

__device__ __forceinline__ v8f wmma_bf(v16bf a, v16bf b, v8f c) {
  return __builtin_amdgcn_wmma_f32_16x16x32_bf16(false, a, false, b, (short)0, c, false, false);
}

#if __has_builtin(__builtin_amdgcn_tensor_load_to_lds) && __has_builtin(__builtin_amdgcn_s_wait_tensorcnt)
#define USE_TDM 1
// 2-D tile DMA: rows x (dim0u * 8B) from global -> LDS, row stride = strideu * 8B.
// Descriptor layout per CDNA5 ISA (D# groups 0/1; groups 2/3 zero => 2-D).
// This toolchain exposes the 6-arg builtin: (v4u g0, v8i g1, v4i g2, v4i g3, v8i extra, i32 cpol).
__device__ __forceinline__ void tdm_load_tile(const bf16* gsrc, unsigned int lds_off,
                                              unsigned int rows, unsigned int dim0u,
                                              unsigned int strideu) {
  unsigned long long ga = (unsigned long long)(size_t)gsrc;
  v4u g0;
  g0[0] = 1u;                                            // count=1 (valid user D#)
  g0[1] = lds_off;                                       // lds_addr
  g0[2] = (unsigned int)ga;                              // global_addr[31:0]
  g0[3] = (unsigned int)((ga >> 32) & 0x01FFFFFFu) | 0x80000000u; // ga[56:32] | type=2
  v8i g1;
  g1[0] = (int)(3u << 16);                               // wg_mask=0, data_size=3 (8B)
  g1[1] = (int)((dim0u & 0xFFFFu) << 16);                // tensor_dim0[15:0]
  g1[2] = (int)(((dim0u >> 16) & 0xFFFFu) | ((rows & 0xFFFFu) << 16));  // d0 hi | tensor_dim1 lo
  g1[3] = (int)(((rows >> 16) & 0xFFFFu) | ((dim0u & 0xFFFFu) << 16));  // d1 hi | tile_dim0
  g1[4] = (int)(rows & 0xFFFFu);                         // tile_dim1 | tile_dim2=0
  g1[5] = (int)strideu;                                  // tensor_dim0_stride[31:0]
  g1[6] = 0;                                             // stride[47:32] | dim1_stride lo
  g1[7] = 0;
  v4i z4 = {0, 0, 0, 0};
  v8i z8 = {0, 0, 0, 0, 0, 0, 0, 0};
  __builtin_amdgcn_tensor_load_to_lds(g0, g1, z4, z4, z8, 0);
}
#endif

// ---------------------------------------------------------------- f32 -> bf16
__global__ void cvt_bf16(const float* __restrict__ src, bf16* __restrict__ dst, int n) {
  int i = blockIdx.x * blockDim.x + threadIdx.x;
  if (i < n) dst[i] = tobf(src[i]);
}

// ------------------------------------------- LN1 + roll(-SHIFT) + window part
__global__ __launch_bounds__(128)
void ln1_shift_part(const float* __restrict__ x, const float* __restrict__ w,
                    const float* __restrict__ b, bf16* __restrict__ hw) {
  int t = blockIdx.x, c = threadIdx.x;
  int bb = t / (HS_ * WS_); int ij = t % (HS_ * WS_);
  int i = ij / WS_, j = ij % WS_;
  int si = (i + SHIFT_) % HS_, sj = (j + SHIFT_) % WS_;
  float v = x[(((size_t)bb * HS_ + si) * WS_ + sj) * C_ + c];
  __shared__ float sb[128];
  sb[c] = v; __syncthreads();
  for (int s = 64; s > 0; s >>= 1) { if (c < s) sb[c] += sb[c + s]; __syncthreads(); }
  float mean = sb[0] * (1.0f / C_); __syncthreads();
  float d = v - mean;
  sb[c] = d * d; __syncthreads();
  for (int s = 64; s > 0; s >>= 1) { if (c < s) sb[c] += sb[c + s]; __syncthreads(); }
  float o = d * rsqrtf(sb[0] * (1.0f / C_) + 1e-5f) * w[c] + b[c];
  int wi = bb * NW_ + (i / WIN_) * NWH_ + (j / WIN_);
  int n  = (i % WIN_) * WIN_ + (j % WIN_);
  hw[((size_t)wi * N_ + n) * C_ + c] = tobf(o);
}

// -------------------------------------------------- LN over C=128 per token
__global__ __launch_bounds__(128)
void ln_token(const float* __restrict__ xin, const float* __restrict__ w,
              const float* __restrict__ b, bf16* __restrict__ out) {
  int t = blockIdx.x, c = threadIdx.x;
  float v = xin[(size_t)t * C_ + c];
  __shared__ float sb[128];
  sb[c] = v; __syncthreads();
  for (int s = 64; s > 0; s >>= 1) { if (c < s) sb[c] += sb[c + s]; __syncthreads(); }
  float mean = sb[0] * (1.0f / C_); __syncthreads();
  float d = v - mean;
  sb[c] = d * d; __syncthreads();
  for (int s = 64; s > 0; s >>= 1) { if (c < s) sb[c] += sb[c + s]; __syncthreads(); }
  float o = d * rsqrtf(sb[0] * (1.0f / C_) + 1e-5f) * w[c] + b[c];
  out[(size_t)t * C_ + c] = tobf(o);
}

// ---------------------------------------------------------------- WMMA GEMM
// out[M][N] = A[M][K](bf16) @ W[N][K]^T(bf16) (+bias) ; K multiple of 128.
// A tile (64 x 128) staged global->LDS by the Tensor Data Mover (wave 0 issues,
// s_wait_tensorcnt, barrier); per-thread fallback if the builtin is absent.
// EPI: 0 bias->bf16 | 1 bias+GELU->bf16 | 2 bias+resid->f32 | 3 bias?->f32
template<int EPI>
__global__ __launch_bounds__(256)
void gemm_bf16(const bf16* __restrict__ A, const bf16* __restrict__ W,
               const float* __restrict__ bias, const float* __restrict__ resid,
               void* __restrict__ outp, int M, int N, int K) {
  __shared__ __align__(128) bf16 As[64 * 128];
  int tid = threadIdx.x;
  int wid = tid >> 5, lane = tid & 31;
  int m0 = blockIdx.x * 64;
  int n0 = blockIdx.y * 128;
  int ln15 = lane & 15;
  int klo    = (lane & 16) ? 8  : 0;   // unused
  int kb_off = (lane & 16) ? 16 : 0;   // B element base
  int nco = n0 + wid * 16 + ln15;

  v8f acc[4];
#pragma unroll
  for (int t = 0; t < 4; ++t)
#pragma unroll
    for (int i = 0; i < 8; ++i) acc[t][i] = 0.0f;

  for (int k0 = 0; k0 < K; k0 += 128) {
#ifdef USE_TDM
    if (wid == 0) {
      tdm_load_tile(A + (size_t)m0 * K + k0, (unsigned int)(size_t)As,
                    64u, 32u /*256B rows in 8B units*/, (unsigned int)(K >> 2));
      __builtin_amdgcn_s_wait_tensorcnt(0);
    }
    __syncthreads();
#else
    {
      int row = tid >> 2;
      int gm  = m0 + row;
      const bf16* src = A + (size_t)gm * K + k0 + (tid & 3) * 32;
      bf16* dstp = As + row * 128 + (tid & 3) * 32;
#pragma unroll
      for (int q = 0; q < 4; ++q) {
        v8bf av;
        if (gm < M) av = *(const v8bf*)(src + q * 8);
        else {
#pragma unroll
          for (int e = 0; e < 8; ++e) av[e] = (bf16)0.0f;
        }
        *(v8bf*)(dstp + q * 8) = av;
      }
    }
    __syncthreads();
#endif
#pragma unroll
    for (int ks = 0; ks < 128; ks += 32) {
      v16bf bfrag = *(const v16bf*)(W + (size_t)nco * K + k0 + ks + kb_off);
#pragma unroll
      for (int t = 0; t < 4; ++t) {
        int r = t * 16 + ln15;
        V16U ua;
        ua.h[0] = *(const v8bf*)(As + r * 128 + ks + kb_off);
        ua.h[1] = *(const v8bf*)(As + r * 128 + ks + kb_off + 8);
        acc[t] = wmma_bf(ua.v, bfrag, acc[t]);
      }
    }
    __syncthreads();
  }

  float bv = bias ? bias[nco] : 0.0f;
  int rbase = (lane & 16) ? 8 : 0;
#pragma unroll
  for (int t = 0; t < 4; ++t) {
#pragma unroll
    for (int i = 0; i < 8; ++i) {
      int m = m0 + t * 16 + i + rbase;
      if (m >= M) continue;
      float v = acc[t][i] + bv;
      size_t oidx = (size_t)m * N + nco;
      if (EPI == 0) {
        ((bf16*)outp)[oidx] = tobf(v);
      } else if (EPI == 1) {
        float g = 0.5f * v * (1.0f + erff(v * 0.70710678118654752f));
        ((bf16*)outp)[oidx] = tobf(g);
      } else if (EPI == 2) {
        ((float*)outp)[oidx] = v + resid[oidx];
      } else {
        ((float*)outp)[oidx] = v;
      }
    }
  }
}

// ----------------------------------------------- fused window attention (WMMA)
// 1 block = 1 window (49 tokens padded to 64), 1 wave = 1 head (DH=32).
__global__ __launch_bounds__(128)
void attn_win(const bf16* __restrict__ qkv, const float* __restrict__ rpb,
              const int* __restrict__ rel, const float* __restrict__ amask,
              bf16* __restrict__ o) {
  __shared__ __align__(32) bf16 P[NH_][64 * 64];
  int win = blockIdx.x;
  int nw  = win & (NW_ - 1);
  int h    = threadIdx.x >> 5;
  int lane = threadIdx.x & 31;
  int ln15 = lane & 15;
  int hiH  = (lane & 16) ? 1 : 0;
  int klo   = hiH ? 8  : 0;
  int dbase = hiH ? 16 : 0;

  const bf16* base = qkv + (size_t)win * (N_ * 3 * C_);
  const bf16* Qp = base + h * DH_;
  const bf16* Kp = base + C_ + h * DH_;
  const bf16* Vp = base + 2 * C_ + h * DH_;

  // ---- S = Q K^T (DH = 32 = one wmma k-step)
  v8f s[4][4];
#pragma unroll
  for (int a = 0; a < 4; ++a)
#pragma unroll
    for (int bq = 0; bq < 4; ++bq)
#pragma unroll
      for (int i = 0; i < 8; ++i) s[a][bq][i] = 0.0f;

  v16bf bk[4];
#pragma unroll
  for (int tj = 0; tj < 4; ++tj) {
    int m = tj * 16 + ln15;
    if (m < N_) bk[tj] = *(const v16bf*)(Kp + (size_t)m * (3 * C_) + dbase);
    else {
#pragma unroll
      for (int e = 0; e < 16; ++e) bk[tj][e] = (bf16)0.0f;
    }
  }
#pragma unroll
  for (int ti = 0; ti < 4; ++ti) {
    int n = ti * 16 + ln15;
    V16U aq;
    if (n < N_) {
      aq.h[0] = *(const v8bf*)(Qp + (size_t)n * (3 * C_) + dbase);
      aq.h[1] = *(const v8bf*)(Qp + (size_t)n * (3 * C_) + dbase + 8);
    } else {
#pragma unroll
      for (int e = 0; e < 16; ++e) aq.v[e] = (bf16)0.0f;
    }
#pragma unroll
    for (int tj = 0; tj < 4; ++tj) s[ti][tj] = wmma_bf(aq.v, bk[tj], s[ti][tj]);
  }

  // ---- scale + rel-pos bias + window mask + softmax (rows split over halves)
  const float scale = 0.17677669529663687f;   // 1/sqrt(32)
#pragma unroll
  for (int ti = 0; ti < 4; ++ti) {
#pragma unroll
    for (int i = 0; i < 8; ++i) {
      int n = ti * 16 + i + hiH * 8;
      float vals[4];
      float mx = -3.0e38f;
#pragma unroll
      for (int tj = 0; tj < 4; ++tj) {
        int m = tj * 16 + ln15;
        float sv;
        if (n < N_ && m < N_)
          sv = s[ti][tj][i] * scale
             + rpb[rel[n * N_ + m] * NH_ + h]
             + amask[((size_t)nw * N_ + n) * N_ + m];
        else sv = -1.0e9f;
        vals[tj] = sv;
        mx = fmaxf(mx, sv);
      }
#pragma unroll
      for (int off = 1; off < 16; off <<= 1) mx = fmaxf(mx, __shfl_xor(mx, off, 32));
      float sum = 0.0f;
#pragma unroll
      for (int tj = 0; tj < 4; ++tj) { vals[tj] = __expf(vals[tj] - mx); sum += vals[tj]; }
#pragma unroll
      for (int off = 1; off < 16; off <<= 1) sum += __shfl_xor(sum, off, 32);
      float inv = 1.0f / sum;
#pragma unroll
      for (int tj = 0; tj < 4; ++tj)
        P[h][n * 64 + tj * 16 + ln15] = tobf(vals[tj] * inv);
    }
  }
  __syncthreads();

  // ---- O = P V  (K = 64 padded tokens -> 2 k-steps)
  v8f oacc[4][2];
#pragma unroll
  for (int a = 0; a < 4; ++a)
#pragma unroll
    for (int bq = 0; bq < 2; ++bq)
#pragma unroll
      for (int i = 0; i < 8; ++i) oacc[a][bq][i] = 0.0f;

#pragma unroll
  for (int kt = 0; kt < 2; ++kt) {
    v16bf bv[2];
    int mb = kt * 32 + dbase;
#pragma unroll
    for (int dj = 0; dj < 2; ++dj) {
      int d = dj * 16 + ln15;
#pragma unroll
      for (int e = 0; e < 16; ++e) {
        int m = mb + e;
        bv[dj][e] = (m < N_) ? Vp[(size_t)m * (3 * C_) + d] : (bf16)0.0f;
      }
    }
#pragma unroll
    for (int ti = 0; ti < 4; ++ti) {
      int r = ti * 16 + ln15;
      V16U ap;
      ap.h[0] = *(const v8bf*)(&P[h][r * 64 + kt * 32 + dbase]);
      ap.h[1] = *(const v8bf*)(&P[h][r * 64 + kt * 32 + dbase + 8]);
#pragma unroll
      for (int dj = 0; dj < 2; ++dj) oacc[ti][dj] = wmma_bf(ap.v, bv[dj], oacc[ti][dj]);
    }
  }

#pragma unroll
  for (int ti = 0; ti < 4; ++ti)
#pragma unroll
    for (int dj = 0; dj < 2; ++dj) {
      int d = dj * 16 + ln15;
#pragma unroll
      for (int i = 0; i < 8; ++i) {
        int n = ti * 16 + i + hiH * 8;
        if (n < N_) o[((size_t)win * N_ + n) * C_ + h * DH_ + d] = tobf(oacc[ti][dj][i]);
      }
    }
}

// ------------------------------- window reverse + roll(+SHIFT) + residual add
__global__ void scatter_resid(const float* __restrict__ x, const float* __restrict__ po,
                              float* __restrict__ x1, int total) {
  int idx = blockIdx.x * blockDim.x + threadIdx.x;
  if (idx >= total) return;
  int c = idx & (C_ - 1);
  int t = idx >> 7;
  int bb = t / (HS_ * WS_); int ij = t % (HS_ * WS_);
  int i = ij / WS_, j = ij % WS_;
  int si = (i + HS_ - SHIFT_) % HS_;
  int sj = (j + WS_ - SHIFT_) % WS_;
  int wi = bb * NW_ + (si / WIN_) * NWH_ + (sj / WIN_);
  int n  = (si % WIN_) * WIN_ + (sj % WIN_);
  x1[idx] = x[idx] + po[((size_t)wi * N_ + n) * C_ + c];
}

// ---------------------------------------------- 2x2 patch gather + LN over 4C
__global__ __launch_bounds__(256)
void merge_ln(const float* __restrict__ x2, const float* __restrict__ w,
              const float* __restrict__ b, bf16* __restrict__ xm) {
  int tok = blockIdx.x;
  int bb = tok / (OH_ * OH_); int r = tok % (OH_ * OH_);
  int oi = r / OH_, oj = r % OH_;
  int tid = threadIdx.x;
  float v[2]; float lsum = 0.0f;
  __shared__ float sb[256];
#pragma unroll
  for (int p = 0; p < 2; ++p) {
    int cc = tid + p * 256;
    int q = cc >> 7; int c = cc & 127;
    int ii = 2 * oi + (q & 1);
    int jj = 2 * oj + (q >> 1);
    v[p] = x2[(((size_t)bb * HS_ + ii) * WS_ + jj) * C_ + c];
    lsum += v[p];
  }
  sb[tid] = lsum; __syncthreads();
  for (int s = 128; s > 0; s >>= 1) { if (tid < s) sb[tid] += sb[tid + s]; __syncthreads(); }
  float mean = sb[0] * (1.0f / 512.0f); __syncthreads();
  float lvar = 0.0f;
#pragma unroll
  for (int p = 0; p < 2; ++p) { float d = v[p] - mean; lvar += d * d; }
  sb[tid] = lvar; __syncthreads();
  for (int s = 128; s > 0; s >>= 1) { if (tid < s) sb[tid] += sb[tid + s]; __syncthreads(); }
  float rinv = rsqrtf(sb[0] * (1.0f / 512.0f) + 1e-5f);
#pragma unroll
  for (int p = 0; p < 2; ++p) {
    int cc = tid + p * 256;
    xm[(size_t)tok * 512 + cc] = tobf((v[p] - mean) * rinv * w[cc] + b[cc]);
  }
}

// ============================================================================
extern "C" void kernel_launch(void* const* d_in, const int* in_sizes, int n_in,
                              void* d_out, int out_size, void* d_ws, size_t ws_size,
                              hipStream_t stream) {
  (void)in_sizes; (void)n_in; (void)out_size; (void)ws_size;
  const float* x       = (const float*)d_in[0];
  const float* n1w     = (const float*)d_in[1];
  const float* n1b     = (const float*)d_in[2];
  const float* qkv_w   = (const float*)d_in[3];
  const float* qkv_b   = (const float*)d_in[4];
  const float* proj_w  = (const float*)d_in[5];
  const float* proj_b  = (const float*)d_in[6];
  const float* rpb     = (const float*)d_in[7];
  const float* n2w     = (const float*)d_in[8];
  const float* n2b     = (const float*)d_in[9];
  const float* fc1_w   = (const float*)d_in[10];
  const float* fc1_b   = (const float*)d_in[11];
  const float* fc2_w   = (const float*)d_in[12];
  const float* fc2_b   = (const float*)d_in[13];
  const float* mnw     = (const float*)d_in[14];
  const float* mnb     = (const float*)d_in[15];
  const float* merge_w = (const float*)d_in[16];
  const int*   rel     = (const int*)d_in[17];
  const float* amask   = (const float*)d_in[18];

  char* ws = (char*)d_ws;
  auto al = [](size_t v) { return (v + 255) & ~(size_t)255; };

  size_t off = 0;
  bf16* qkvw_bf  = (bf16*)(ws + off); off += al((size_t)384 * 128 * 2);
  bf16* projw_bf = (bf16*)(ws + off); off += al((size_t)128 * 128 * 2);
  bf16* fc1w_bf  = (bf16*)(ws + off); off += al((size_t)512 * 128 * 2);
  bf16* fc2w_bf  = (bf16*)(ws + off); off += al((size_t)128 * 512 * 2);
  bf16* mergew_bf = (bf16*)(ws + off); off += al((size_t)256 * 512 * 2);

  // region A: hw + qkv, later aliased by g (FC1 output)
  size_t regionA  = off;
  size_t hw_bytes = al((size_t)NTOK * C_ * 2);
  size_t qkv_bytes = al((size_t)NTOK * 384 * 2 + 64 * 384 * 2);  // pad for m>=49 reads
  bf16* hw_bf  = (bf16*)(ws + regionA);
  bf16* qkvbuf = (bf16*)(ws + regionA + hw_bytes);
  bf16* g_bf   = (bf16*)(ws + regionA);                 // alias (hw/qkv dead by FC1)
  size_t gbytes = al((size_t)NTOK * HID_ * 2);
  size_t regA_sz = hw_bytes + qkv_bytes; if (gbytes > regA_sz) regA_sz = gbytes;

  // region B: o + po, later aliased by xm + x2
  size_t regionB = regionA + regA_sz;
  size_t o_bytes  = al((size_t)NTOK * C_ * 2);
  size_t po_bytes = al((size_t)NTOK * C_ * 4);
  bf16*  o_bf  = (bf16*)(ws + regionB);
  float* po    = (float*)(ws + regionB + o_bytes);
  bf16*  xm_bf = (bf16*)(ws + regionB);                 // alias o (dead after proj)
  float* x2    = (float*)(ws + regionB + o_bytes);      // alias po (dead after scatter)
  size_t oc = regionB + o_bytes + po_bytes;

  float* x1   = (float*)(ws + oc); oc += al((size_t)NTOK * C_ * 4);
  bf16*  h2_bf = (bf16*)(ws + oc); oc += al((size_t)NTOK * C_ * 2);

  // ---- weight conversions
  cvt_bf16<<<(384 * 128 + 255) / 256, 256, 0, stream>>>(qkv_w, qkvw_bf, 384 * 128);
  cvt_bf16<<<(128 * 128 + 255) / 256, 256, 0, stream>>>(proj_w, projw_bf, 128 * 128);
  cvt_bf16<<<(512 * 128 + 255) / 256, 256, 0, stream>>>(fc1_w, fc1w_bf, 512 * 128);
  cvt_bf16<<<(128 * 512 + 255) / 256, 256, 0, stream>>>(fc2_w, fc2w_bf, 128 * 512);
  cvt_bf16<<<(256 * 512 + 255) / 256, 256, 0, stream>>>(merge_w, mergew_bf, 256 * 512);

  // ---- LN1 + shift + window partition
  ln1_shift_part<<<NTOK, 128, 0, stream>>>(x, n1w, n1b, hw_bf);

  // ---- QKV GEMM: [200704,128] @ [384,128]^T
  gemm_bf16<0><<<dim3(NTOK / 64, 3), 256, 0, stream>>>(hw_bf, qkvw_bf, qkv_b, nullptr,
                                                       qkvbuf, NTOK, 384, 128);
  // ---- attention per window
  attn_win<<<NWINT, 128, 0, stream>>>(qkvbuf, rpb, rel, amask, o_bf);

  // ---- proj GEMM -> f32
  gemm_bf16<3><<<dim3(NTOK / 64, 1), 256, 0, stream>>>(o_bf, projw_bf, proj_b, nullptr,
                                                       po, NTOK, 128, 128);
  // ---- reverse windows + roll + residual
  scatter_resid<<<(NTOK * C_ + 255) / 256, 256, 0, stream>>>(x, po, x1, NTOK * C_);

  // ---- LN2
  ln_token<<<NTOK, 128, 0, stream>>>(x1, n2w, n2b, h2_bf);

  // ---- FC1 + GELU
  gemm_bf16<1><<<dim3(NTOK / 64, 4), 256, 0, stream>>>(h2_bf, fc1w_bf, fc1_b, nullptr,
                                                       g_bf, NTOK, 512, 128);
  // ---- FC2 + residual -> x2
  gemm_bf16<2><<<dim3(NTOK / 64, 1), 256, 0, stream>>>(g_bf, fc2w_bf, fc2_b, x1,
                                                       x2, NTOK, 128, 512);
  // ---- patch-merge gather + LN(512)
  merge_ln<<<MTOK, 256, 0, stream>>>(x2, mnw, mnb, xm_bf);

  // ---- merge GEMM -> d_out [50176,256] f32
  gemm_bf16<3><<<dim3(MTOK / 64, 2), 256, 0, stream>>>(xm_bf, mergew_bf, nullptr, nullptr,
                                                       (float*)d_out, MTOK, 256, 512);
}
